// LSTMAutoencoder_85633057948376
// MI455X (gfx1250) — matrix operand order repaired and checked by reading, not compiled, not run on hardware
//
#include <hip/hip_runtime.h>

// LSTM autoencoder: I=32, H=64, B=4096, T=256.
// Persistent-per-block recurrence: 128 blocks x 32 batch rows, 256 threads (8 wave32).
// Weights resident in LDS (f16), cell state c and gate biases in WMMA accumulator
// registers, h ping-pong in LDS, gates via v_wmma_f32_16x16x32_f16.

#define I_DIM   32
#define H_DIM   64
#define NG      256          // 4*H
#define T_STEPS 256
#define TILE_B  32
#define TI      (T_STEPS * I_DIM)   // 8192
#define XB_SZ   (TILE_B * I_DIM)    // 1024 halves per parity
#define HB_SZ   (TILE_B * H_DIM)    // 2048 halves per parity

typedef __attribute__((ext_vector_type(16))) _Float16 v16h;
typedef __attribute__((ext_vector_type(8)))  _Float16 v8h;
typedef __attribute__((ext_vector_type(8)))  float    v8f;

static __device__ __forceinline__ float fast_exp(float x) {
    return __builtin_amdgcn_exp2f(x * 1.44269504089f);
}
static __device__ __forceinline__ float sigf(float x) {
    return __builtin_amdgcn_rcpf(1.0f + fast_exp(-x));   // saturates at +-inf
}
static __device__ __forceinline__ float tanh_(float x) {
    return 1.0f - 2.0f * __builtin_amdgcn_rcpf(1.0f + fast_exp(2.0f * x));
}
static __device__ __forceinline__ v8f bcast8(float v) {
    v8f r;
#pragma unroll
    for (int i = 0; i < 8; ++i) r[i] = v;
    return r;
}
static __device__ __forceinline__ v8f wmma_f16(v16h a, v16h b, v8f c) {
    // D = A(16x32 f16) * B(32x16 f16) + C(f32); SRC2 and VDST are distinct,
    // so passing a persistent bias vector as C costs no copies.
    return __builtin_amdgcn_wmma_f32_16x16x32_f16(false, a, false, b, (short)0, c, false, false);
}

// A fragment: act row-major [TILE_B][stride] f16 in LDS. Rows row0..row0+15,
// K-chunk [kb,kb+32). Lane m=lane&15, g=lane>>4: halves 0..7 = K kb+8g..+7,
// halves 8..15 = K kb+16+8g..+7.
static __device__ __forceinline__ v16h a_frag(const _Float16* __restrict__ act,
                                              int row0, int stride, int kb, int lane) {
    const int m = lane & 15, g = lane >> 4;
    const _Float16* p = act + (row0 + m) * stride + kb + 8 * g;
    v8h lo = *(const v8h*)(p);
    v8h hi = *(const v8h*)(p + 16);
    v16h a;
#pragma unroll
    for (int i = 0; i < 8; ++i) { a[i] = lo[i]; a[i + 8] = hi[i]; }
    return a;
}

// B fragment for gates = act * W^T, W row-major [N][K] f16 in LDS.
// B[k][n] = W[n][k]; lane n=lane&15, g=lane>>4: halves 0..7 = k kb+8g..+7,
// halves 8..15 = k kb+16+8g..+7 of weight row n.
static __device__ __forceinline__ v16h b_frag(const _Float16* __restrict__ W,
                                              int nb, int K, int kb, int lane) {
    const int n = lane & 15, g = lane >> 4;
    const _Float16* p = W + (nb + n) * K + kb + 8 * g;
    v8h lo = *(const v8h*)(p);
    v8h hi = *(const v8h*)(p + 16);
    v16h b;
#pragma unroll
    for (int i = 0; i < 8; ++i) { b[i] = lo[i]; b[i + 8] = hi[i]; }
    return b;
}

// One LSTM cell step for this wave's 16x16 (rows row0.., h-cols q*16..) block.
// Gate n-tiles for h-col tile q are {q, q+4, q+8, q+12} (i,f,g,o).
// biasv[4] are persistent splatted bias vectors used directly as WMMA C input.
template <int KX>
static __device__ __forceinline__ void lstm_tile(
    const _Float16* __restrict__ actX,          // [TILE_B][KX]
    const _Float16* __restrict__ actH,          // [TILE_B][H]
    const _Float16* __restrict__ Wih,           // [NG][KX]
    const _Float16* __restrict__ Whh,           // [NG][H]
    const v8f (&biasv)[4],
    _Float16*       __restrict__ hOut,          // [TILE_B][H]
    v8f& c, int row0, int q, int lane)
{
    const int n = lane & 15, g = lane >> 4;
    v8f acc[4];
#pragma unroll
    for (int gi = 0; gi < 4; ++gi) {
        const int nb = (q + 4 * gi) * 16;
        v8f a = wmma_f16(a_frag(actX, row0, KX, 0, lane),
                         b_frag(Wih, nb, KX, 0, lane), biasv[gi]);
#pragma unroll
        for (int kb = 32; kb < KX; kb += 32)
            a = wmma_f16(a_frag(actX, row0, KX, kb, lane),
                         b_frag(Wih, nb, KX, kb, lane), a);
        a = wmma_f16(a_frag(actH, row0, H_DIM, 0,  lane), b_frag(Whh, nb, H_DIM, 0,  lane), a);
        a = wmma_f16(a_frag(actH, row0, H_DIM, 32, lane), b_frag(Whh, nb, H_DIM, 32, lane), a);
        acc[gi] = a;
    }
#pragma unroll
    for (int r = 0; r < 8; ++r) {
        float iv = sigf(acc[0][r]);
        float fv = sigf(acc[1][r]);
        float gv = tanh_(acc[2][r]);
        float ov = sigf(acc[3][r]);
        float cn = fv * c[r] + iv * gv;
        c[r] = cn;
        float hv = ov * tanh_(cn);
        hOut[(row0 + r + 8 * g) * H_DIM + q * 16 + n] = (_Float16)hv;
    }
}

static __device__ __forceinline__ void cvt16(_Float16* __restrict__ dst,
                                             const float* __restrict__ src,
                                             int nelem, int tid) {
    for (int i = tid; i < nelem; i += 256) dst[i] = (_Float16)src[i];
}

static __device__ __forceinline__ void load_x(_Float16* __restrict__ dst,
                                              const float* __restrict__ x,
                                              int b0, int t, int tid) {
    for (int i = tid; i < XB_SZ; i += 256) {
        const int r = i >> 5, cc = i & 31;
        dst[i] = (_Float16)x[(size_t)(b0 + r) * TI + t * I_DIM + cc];
    }
}

// LDS byte offsets (16B aligned). Total 139264 B (< 160 KB => 2 WG/WGP possible).
#define OFF_WIH0   0        // 256x32 f16
#define OFF_WHH0   16384    // 256x64 f16
#define OFF_WIH1   49152    // 256x64 f16
#define OFF_WHH1   81920    // 256x64 f16
#define OFF_OUTW   114688   // 32x64 f16
#define OFF_XB     118784   // 2 x 32x32 f16 (ping-pong, encoder prefetch)
#define OFF_H0     122880   // 2 x 32x64 f16 (ping-pong)
#define OFF_H1     131072   // 2 x 32x64 f16
#define SMEM_BYTES 139264

__global__ __launch_bounds__(256, 1)
void lstm_ae_kernel(const float* __restrict__ x,
                    const float* eW0, const float* eU0, const float* eb0a, const float* eb0b,
                    const float* eW1, const float* eU1, const float* eb1a, const float* eb1b,
                    const float* dW0, const float* dU0, const float* db0a, const float* db0b,
                    const float* dW1, const float* dU1, const float* db1a, const float* db1b,
                    const float* oW, const float* ob, const float* stok,
                    float* __restrict__ out)
{
    extern __shared__ __align__(16) char smem[];
    _Float16* Wih0 = (_Float16*)(smem + OFF_WIH0);
    _Float16* Whh0 = (_Float16*)(smem + OFF_WHH0);
    _Float16* Wih1 = (_Float16*)(smem + OFF_WIH1);
    _Float16* Whh1 = (_Float16*)(smem + OFF_WHH1);
    _Float16* outW = (_Float16*)(smem + OFF_OUTW);
    _Float16* xb   = (_Float16*)(smem + OFF_XB);
    _Float16* h0b  = (_Float16*)(smem + OFF_H0);
    _Float16* h1b  = (_Float16*)(smem + OFF_H1);

    const int tid  = threadIdx.x;
    const int wave = tid >> 5;          // 0..7
    const int lane = tid & 31;
    const int q    = wave & 3;          // h-column tile (16 cols)
    const int row0 = (wave >> 2) * 16;  // row tile base (0 or 16)
    const int b0   = blockIdx.x * TILE_B;
    const int ln   = lane & 15;         // column within tile

    // ---- encoder weights -> LDS; biases -> persistent splatted registers ----
    cvt16(Wih0, eW0, NG * I_DIM, tid);
    cvt16(Whh0, eU0, NG * H_DIM, tid);
    cvt16(Wih1, eW1, NG * H_DIM, tid);
    cvt16(Whh1, eU1, NG * H_DIM, tid);
    for (int i = tid; i < 2 * HB_SZ; i += 256) { h0b[i] = (_Float16)0.f; h1b[i] = (_Float16)0.f; }

    v8f bias0v[4], bias1v[4];
#pragma unroll
    for (int gi = 0; gi < 4; ++gi) {
        const int idx = (q + 4 * gi) * 16 + ln;
        bias0v[gi] = bcast8(eb0a[idx] + eb0b[idx]);
        bias1v[gi] = bcast8(eb1a[idx] + eb1b[idx]);
    }
    v8f c0 = bcast8(0.f), c1 = bcast8(0.f);

    load_x(xb, x, b0, 0, tid);          // preload t=0 into parity 0
    __syncthreads();

    // ---- encoder: 256 sequential steps; x(t+1) prefetched during compute ----
    for (int t = 0; t < T_STEPS; ++t) {
        const int p = t & 1;
        if (t + 1 < T_STEPS)            // parity 1-p was last read two barriers ago
            load_x(xb + (1 - p) * XB_SZ, x, b0, t + 1, tid);
        lstm_tile<I_DIM>(xb + p * XB_SZ, h0b + p * HB_SZ, Wih0, Whh0, bias0v,
                         h0b + (1 - p) * HB_SZ, c0, row0, q, lane);
        __syncthreads();
        lstm_tile<H_DIM>(h0b + (1 - p) * HB_SZ, h1b + p * HB_SZ, Wih1, Whh1, bias1v,
                         h1b + (1 - p) * HB_SZ, c1, row0, q, lane);
        __syncthreads();
    }
    // final h after t=T-1 lives in parity 0 => decoder reads parity t&1 from 0.

    // ---- swap in decoder weights/biases; init feedback input ----
    cvt16(Wih0, dW0, NG * I_DIM, tid);
    cvt16(Whh0, dU0, NG * H_DIM, tid);
    cvt16(Wih1, dW1, NG * H_DIM, tid);
    cvt16(Whh1, dU1, NG * H_DIM, tid);
    cvt16(outW, oW, I_DIM * H_DIM, tid);
#pragma unroll
    for (int gi = 0; gi < 4; ++gi) {
        const int idx = (q + 4 * gi) * 16 + ln;
        bias0v[gi] = bcast8(db0a[idx] + db0b[idx]);
        bias1v[gi] = bcast8(db1a[idx] + db1b[idx]);
    }
    v8f pb = bcast8(0.f);
    if (wave < 4) pb = bcast8(ob[(wave & 1) * 16 + ln]);   // projection bias
    for (int i = tid; i < XB_SZ; i += 256) xb[i] = (_Float16)stok[i & (I_DIM - 1)];
    __syncthreads();

    // ---- decoder: 256 steps with WMMA output projection + feedback ----
    for (int t = 0; t < T_STEPS; ++t) {
        const int p = t & 1;
        lstm_tile<I_DIM>(xb, h0b + p * HB_SZ, Wih0, Whh0, bias0v,
                         h0b + (1 - p) * HB_SZ, c0, row0, q, lane);
        __syncthreads();
        lstm_tile<H_DIM>(h0b + (1 - p) * HB_SZ, h1b + p * HB_SZ, Wih1, Whh1, bias1v,
                         h1b + (1 - p) * HB_SZ, c1, row0, q, lane);
        __syncthreads();
        // out = h1 @ out_W^T + out_b : 32x32, 4 tiles on waves 0..3 (EXEC all-ones per wave)
        if (wave < 4) {
            const _Float16* h1cur = h1b + (1 - p) * HB_SZ;
            const int prow0 = (wave >> 1) * 16, pnb = (wave & 1) * 16;
            const int g = lane >> 4;
            v8f a = wmma_f16(a_frag(h1cur, prow0, H_DIM, 0, lane),
                             b_frag(outW, pnb, H_DIM, 0, lane), pb);
            a = wmma_f16(a_frag(h1cur, prow0, H_DIM, 32, lane),
                         b_frag(outW, pnb, H_DIM, 32, lane), a);
#pragma unroll
            for (int r = 0; r < 8; ++r) {
                const int m = prow0 + r + 8 * g;
                const float v = a[r];
                out[(size_t)(b0 + m) * TI + t * I_DIM + pnb + ln] = v;  // f32 result
                xb[m * I_DIM + pnb + ln] = (_Float16)v;                 // feedback input
            }
        }
        __syncthreads();
    }
}

extern "C" void kernel_launch(void* const* d_in, const int* in_sizes, int n_in,
                              void* d_out, int out_size, void* d_ws, size_t ws_size,
                              hipStream_t stream) {
    (void)in_sizes; (void)n_in; (void)d_ws; (void)ws_size; (void)out_size;
    const float* p[20];
    for (int i = 0; i < 20; ++i) p[i] = (const float*)d_in[i];
    dim3 grid(4096 / TILE_B);   // 128 persistent blocks
    dim3 block(256);            // 8 wave32
    hipLaunchKernelGGL(lstm_ae_kernel, grid, block, SMEM_BYTES, stream,
                       p[0],
                       p[1], p[2], p[3], p[4],
                       p[5], p[6], p[7], p[8],
                       p[9], p[10], p[11], p[12],
                       p[13], p[14], p[15], p[16],
                       p[17], p[18], p[19],
                       (float*)d_out);
}
